// Correlation_3350074491441
// MI455X (gfx1250) — matrix operand order repaired and checked by reading, not compiled, not run on hardware
//
#include <hip/hip_runtime.h>

// FlowNetC correlation: B=8, C=256, H=48, W=64, 441 displacement channels,
// stride2=2, pad=20. out[b, i*21+j, y, x] =
//   (1/256) * sum_c in1[b,c,y,x] * in2[b,c, y+2i-20, x+2j-20]   (zero-padded)
//
// Pass 1: convert+transpose inputs to channel-last bf16 in d_ws.
// Pass 2: band-GEMM via V_WMMA_F32_16X16X32_BF16. A-fragments live in VGPRs
//         for the whole kernel; B tiles double-buffered in LDS and staged with
//         GLOBAL_LOAD_ASYNC_TO_LDS_B128 (ASYNCcnt) overlapped with WMMAs.

#define BDIM  256
#define C_CH  256
#define Hh    48
#define Ww    64
#define NB    8
#define GRID  (NB * Hh)   // 384 (b,y) rows
#define GW    21          // displacements per axis
#define AP    264         // transpose-kernel LDS pitch (bf16), rows 16B aligned
#define BP    136         // B LDS pitch (bf16): 128 + 8 pad, rows 16B aligned
#define BU    112         // B u-rows (WMMA reads u in [0,112))

typedef __bf16 bf16;
typedef bf16  v16bf __attribute__((ext_vector_type(16)));
typedef bf16  v8bf  __attribute__((ext_vector_type(8)));
typedef float v8f   __attribute__((ext_vector_type(8)));

__device__ __forceinline__ bf16 f2bf(float f) {
  // round-to-nearest-even fp32 -> bf16
  unsigned u = __builtin_bit_cast(unsigned, f);
  u = u + 0x7FFFu + ((u >> 16) & 1u);
  unsigned short s = (unsigned short)(u >> 16);
  return __builtin_bit_cast(bf16, s);
}

__device__ __forceinline__ v16bf ld_frag(const bf16* p0, const bf16* p1) {
  v8bf lo = *(const v8bf*)p0;   // b128 load (16B aligned; LDS or global)
  v8bf hi = *(const v8bf*)p1;
  v16bf r;
#pragma unroll
  for (int t = 0; t < 8; ++t) { r[t] = lo[t]; r[t + 8] = hi[t]; }
  return r;
}

// One async 16B global->LDS copy (GV addressing, tracked on ASYNCcnt).
__device__ __forceinline__ void async_b128(unsigned lds_off, const void* gaddr) {
  unsigned long long ga = (unsigned long long)gaddr;
  asm volatile("global_load_async_to_lds_b128 %0, %1, off"
               :: "v"(lds_off), "v"(ga) : "memory");
}
__device__ __forceinline__ void wait_async0() {
  asm volatile("s_wait_asynccnt 0" ::: "memory");
}
__device__ __forceinline__ unsigned lds_off32(const void* p) {
  // generic shared pointer: low 32 bits are the LDS byte address
  return (unsigned)(unsigned long long)p;
}

// ---------------------------------------------------------------------------
// Pass 1: [B,C,H,W] f32  ->  [B,H,W,C] bf16 (channel-last), LDS transpose.
// blockIdx.x in [0, 2*GRID): first half converts in1, second half in2.
// ---------------------------------------------------------------------------
__global__ __launch_bounds__(BDIM)
void convert_transpose(const float* __restrict__ in1,
                       const float* __restrict__ in2,
                       bf16* __restrict__ out1,
                       bf16* __restrict__ out2) {
  __shared__ bf16 T[64 * AP];   // 33,792 B

  const int row = blockIdx.x % GRID;
  const float* src = (blockIdx.x < GRID) ? in1 : in2;
  bf16*        dst = (blockIdx.x < GRID) ? out1 : out2;
  const int b = row / Hh, y = row % Hh;

  for (int t = threadIdx.x; t < 64 * C_CH; t += BDIM) {   // coalesced along x
    int x = t & 63, c = t >> 6;
    T[x * AP + c] = f2bf(src[(((size_t)b * C_CH + c) * Hh + y) * Ww + x]);
  }
  __syncthreads();
  char* drow = (char*)(dst + (size_t)row * Ww * C_CH);
  for (int q = threadIdx.x; q < 64 * 32; q += BDIM) {     // coalesced along c
    int x = q >> 5, seg = q & 31;
    *(uint4*)(drow + x * 512 + seg * 16) =
        *(const uint4*)((const char*)&T[x * AP] + seg * 16);
  }
}

// ---------------------------------------------------------------------------
// Pass 2: correlation, A in registers, double-buffered async B staging.
// ---------------------------------------------------------------------------
__global__ __launch_bounds__(BDIM)
void corr_wmma_async(const bf16* __restrict__ in1t,
                     const bf16* __restrict__ in2t,
                     float* __restrict__ out) {
  __shared__ bf16 B0_lds[BU * BP];   // 30,464 B
  __shared__ bf16 B1_lds[BU * BP];   // 30,464 B  -> total 60,928 B

  const int tid  = threadIdx.x;
  const int b    = blockIdx.x / Hh;
  const int y    = blockIdx.x % Hh;

  const int wave = tid >> 5;
  const int lane = tid & 31;
  const int lh   = lane >> 4;
  const int ln   = lane & 15;
  const int mt   = wave >> 1;               // M tile 0..3
  const int ut0  = mt + 2 * (wave & 1);     // U tiles (ut0, ut0+1)

  // ---- A fragments: 8 K-steps, held in VGPRs for the whole kernel ----
  v16bf afrag[8];
  {
    const bf16* arow =
        in1t + ((size_t)(b * Hh + y) * Ww + (mt * 16 + ln)) * C_CH;
#pragma unroll
    for (int k8 = 0; k8 < 8; ++k8) {
      const bf16* p = arow + k8 * 32 + lh * 16;   // 16B-aligned global runs
      afrag[k8] = ld_frag(p, p + 8);
    }
  }
  // ---- Zero rows u<16 and u>=96 of BOTH buffers, once (u 16..19, 84..95 not zeroed) ----
  for (int t = tid; t < 2 * 48 * 128; t += BDIM) {
    bf16* base = (t < 48 * 128) ? B0_lds : B1_lds;
    int tt = t & (48 * 128 - 1);
    int ui = tt >> 7;
    int u  = (ui < 20) ? ui : (ui + 64);
    base[u * BP + (tt & 127)] = f2bf(0.0f);
  }
  __syncthreads();

  const bf16* B0b0 = &B0_lds[(ut0 * 16 + ln) * BP];
  const bf16* B0b1 = B0b0 + 16 * BP;
  const bf16* B1b0 = &B1_lds[(ut0 * 16 + ln) * BP];
  const bf16* B1b1 = B1b0 + 16 * BP;
  const unsigned off0 = lds_off32(B0_lds);
  const unsigned off1 = lds_off32(B1_lds);
  const float inv = 1.0f / 256.0f;

  for (int i = 0; i < GW; ++i) {
    const int r = y + 2 * i - 20;   // in2 source row (uniform across block)
    if (r < 0 || r >= Hh) {
      for (int t = tid; t < GW * Ww; t += BDIM) {
        int j = t >> 6, x = t & 63;
        out[(((size_t)b * 441 + i * GW + j) * Hh + y) * Ww + x] = 0.0f;
      }
      continue;
    }

    const char* brow = (const char*)(in2t + (size_t)(b * Hh + r) * Ww * C_CH);

    // stage chunk0 (channels 0..127) -> B0 ; rows u = x2+20
#pragma unroll
    for (int k = 0; k < 4; ++k) {
      int q = tid + k * BDIM;            // 1024 = 4*256 chunks of 16B
      int x2 = q >> 4, seg = q & 15;
      async_b128(off0 + (x2 + 20) * (BP * 2) + seg * 16,
                 brow + x2 * 512 + seg * 16);
    }
    wait_async0();
    __syncthreads();                     // chunk0 ready; B1 reads all done

    // stage chunk1 (channels 128..255) -> B1, overlapped with chunk0 WMMAs
#pragma unroll
    for (int k = 0; k < 4; ++k) {
      int q = tid + k * BDIM;
      int x2 = q >> 4, seg = q & 15;
      async_b128(off1 + (x2 + 20) * (BP * 2) + seg * 16,
                 brow + x2 * 512 + 256 + seg * 16);
    }

    v8f acc0 = {}; v8f acc1 = {};
#pragma unroll
    for (int kk = 0; kk < 4; ++kk) {     // chunk0 WMMAs (B0, afrag[0..3])
      const int kB = kk * 32;
      v16bf b0 = ld_frag(B0b0 + kB + lh * 16, B0b0 + kB + lh * 16 + 8);
      v16bf b1 = ld_frag(B0b1 + kB + lh * 16, B0b1 + kB + lh * 16 + 8);
      acc0 = __builtin_amdgcn_wmma_f32_16x16x32_bf16(
          false, afrag[kk], false, b0, (short)0, acc0, false, false);
      acc1 = __builtin_amdgcn_wmma_f32_16x16x32_bf16(
          false, afrag[kk], false, b1, (short)0, acc1, false, false);
    }
    wait_async0();
    __syncthreads();                     // chunk1 ready; B0 reads all done

#pragma unroll
    for (int kk = 0; kk < 4; ++kk) {     // chunk1 WMMAs (B1, afrag[4..7])
      const int kB = kk * 32;
      v16bf b0 = ld_frag(B1b0 + kB + lh * 16, B1b0 + kB + lh * 16 + 8);
      v16bf b1 = ld_frag(B1b1 + kB + lh * 16, B1b1 + kB + lh * 16 + 8);
      acc0 = __builtin_amdgcn_wmma_f32_16x16x32_bf16(
          false, afrag[4 + kk], false, b0, (short)0, acc0, false, false);
      acc1 = __builtin_amdgcn_wmma_f32_16x16x32_bf16(
          false, afrag[4 + kk], false, b1, (short)0, acc1, false, false);
    }

    // ---- Extract band entries straight from accumulators ----
    // C layout: vector elem v <-> M = 16*mt + v + 8*lh, N(=u) = 16*ut + ln
#pragma unroll
    for (int v = 0; v < 8; ++v) {
      const int m = mt * 16 + v + lh * 8;     // output x
      {
        int u = ut0 * 16 + ln, d = u - m;
        if (d >= 0 && d <= 40 && !(d & 1)) {
          int j = d >> 1;
          out[(((size_t)b * 441 + i * GW + j) * Hh + y) * Ww + m] = acc0[v] * inv;
        }
      }
      {
        int u = (ut0 + 1) * 16 + ln, d = u - m;
        if (d >= 0 && d <= 40 && !(d & 1)) {
          int j = d >> 1;
          out[(((size_t)b * 441 + i * GW + j) * Hh + y) * Ww + m] = acc1[v] * inv;
        }
      }
    }
  }
}

// ---------------------------------------------------------------------------
// Fallback (no workspace): fused staging with on-the-fly f32->bf16 convert.
// ---------------------------------------------------------------------------
__global__ __launch_bounds__(BDIM)
void corr_wmma_legacy(const float* __restrict__ in1,
                      const float* __restrict__ in2,
                      float* __restrict__ out) {
  __shared__ bf16 A_lds[64 * AP];
  __shared__ bf16 B_lds[BU * BP];

  const int tid  = threadIdx.x;
  const int b    = blockIdx.x / Hh;
  const int y    = blockIdx.x % Hh;
  const int wave = tid >> 5;
  const int lane = tid & 31;
  const int lh   = lane >> 4;
  const int ln   = lane & 15;
  const int mt   = wave >> 1;
  const int ut0  = mt + 2 * (wave & 1);

  for (int t = tid; t < 64 * C_CH; t += BDIM) {
    int x = t & 63, c = t >> 6;
    A_lds[x * AP + c] = f2bf(in1[(((size_t)b * C_CH + c) * Hh + y) * Ww + x]);
  }
  for (int t = tid; t < 48 * 128; t += BDIM) {
    int ui = t >> 7;
    int u  = (ui < 20) ? ui : (ui + 64);
    B_lds[u * BP + (t & 127)] = f2bf(0.0f);
  }
  __syncthreads();

  const bf16* Abase  = &A_lds[(mt * 16 + ln) * AP];
  const bf16* Bbase0 = &B_lds[(ut0 * 16 + ln) * BP];
  const bf16* Bbase1 = Bbase0 + 16 * BP;
  const float inv = 1.0f / 256.0f;

  for (int i = 0; i < GW; ++i) {
    const int r = y + 2 * i - 20;
    if (r < 0 || r >= Hh) {
      for (int t = tid; t < GW * Ww; t += BDIM) {
        int j = t >> 6, x = t & 63;
        out[(((size_t)b * 441 + i * GW + j) * Hh + y) * Ww + x] = 0.0f;
      }
      continue;
    }
    v8f acc0 = {}; v8f acc1 = {};
    for (int ch = 0; ch < 2; ++ch) {
      for (int t = tid; t < 64 * 128; t += BDIM) {
        int x2 = t & 63, cl = t >> 6;
        B_lds[(x2 + 20) * BP + cl] =
            f2bf(in2[(((size_t)b * C_CH + (ch * 128 + cl)) * Hh + r) * Ww + x2]);
      }
      __syncthreads();
#pragma unroll
      for (int kk = 0; kk < 4; ++kk) {
        const int kA = ch * 128 + kk * 32;
        const int kB = kk * 32;
        v16bf a  = ld_frag(Abase  + kA + lh * 16, Abase  + kA + lh * 16 + 8);
        v16bf b0 = ld_frag(Bbase0 + kB + lh * 16, Bbase0 + kB + lh * 16 + 8);
        v16bf b1 = ld_frag(Bbase1 + kB + lh * 16, Bbase1 + kB + lh * 16 + 8);
        acc0 = __builtin_amdgcn_wmma_f32_16x16x32_bf16(
            false, a, false, b0, (short)0, acc0, false, false);
        acc1 = __builtin_amdgcn_wmma_f32_16x16x32_bf16(
            false, a, false, b1, (short)0, acc1, false, false);
      }
      __syncthreads();
    }
#pragma unroll
    for (int v = 0; v < 8; ++v) {
      const int m = mt * 16 + v + lh * 8;
      {
        int u = ut0 * 16 + ln, d = u - m;
        if (d >= 0 && d <= 40 && !(d & 1)) {
          int j = d >> 1;
          out[(((size_t)b * 441 + i * GW + j) * Hh + y) * Ww + m] = acc0[v] * inv;
        }
      }
      {
        int u = (ut0 + 1) * 16 + ln, d = u - m;
        if (d >= 0 && d <= 40 && !(d & 1)) {
          int j = d >> 1;
          out[(((size_t)b * 441 + i * GW + j) * Hh + y) * Ww + m] = acc1[v] * inv;
        }
      }
    }
  }
}

extern "C" void kernel_launch(void* const* d_in, const int* in_sizes, int n_in,
                              void* d_out, int out_size, void* d_ws, size_t ws_size,
                              hipStream_t stream) {
  (void)in_sizes; (void)n_in; (void)out_size;
  const float* in1 = (const float*)d_in[0];
  const float* in2 = (const float*)d_in[1];
  float* out = (float*)d_out;

  const size_t elems = (size_t)NB * Hh * Ww * C_CH;        // per input
  const size_t need  = 2 * elems * sizeof(bf16);           // 25,165,824 B

  if (ws_size >= need && d_ws != nullptr) {
    bf16* w1 = (bf16*)d_ws;
    bf16* w2 = w1 + elems;
    convert_transpose<<<dim3(2 * GRID), dim3(BDIM), 0, stream>>>(in1, in2, w1, w2);
    corr_wmma_async<<<dim3(GRID), dim3(BDIM), 0, stream>>>(w1, w2, out);
  } else {
    corr_wmma_legacy<<<dim3(GRID), dim3(BDIM), 0, stream>>>(in1, in2, out);
  }
}
